// SynapticLayer_45380624450079
// MI455X (gfx1250) — matrix operand order repaired and checked by reading, not compiled, not run on hardware
//
#include <hip/hip_runtime.h>

#define B_  256
#define T_  500
#define I_  1024
#define O_  2048

#define BT  32       // B rows per workgroup
#define OT  64       // O cols per workgroup
#define KP  1032     // padded LDS k-stride (f16 elems): keeps 16B align, spreads banks
#define PW  68       // padded n-stride (f32) for partial scratch: 272B = 17*16, bank-safe
#define THREADS 256

#define SW_BYTES   (OT * KP * 2)          // 132,096
#define SA_BYTES   (BT * KP * 2)          //  66,048 per buffer
#define SP_BYTES   (4 * BT * PW * 4)      //  34,816
#define SMEM_BYTES (SW_BYTES + 2 * SA_BYTES + SP_BYTES)   // 299,008 < 320KB

typedef __attribute__((ext_vector_type(16))) _Float16 v16h;
typedef __attribute__((ext_vector_type(8)))  float    v8f;
typedef __attribute__((ext_vector_type(4)))  float    nf4;   // native vec4

union Frag16 { v16h v; uint4 q[2]; };

static __device__ __forceinline__ unsigned short f2h(float f) {
  _Float16 h = (_Float16)f;          // v_cvt_f16_f32 (RNE)
  return __builtin_bit_cast(unsigned short, h);
}

// pack two f32 -> two f16 in one instruction (v_cvt_pkrtz_f16_f32)
static __device__ __forceinline__ unsigned int pk2(float a, float b) {
#if __has_builtin(__builtin_amdgcn_cvt_pkrtz)
  return __builtin_bit_cast(unsigned int, __builtin_amdgcn_cvt_pkrtz(a, b));
#else
  return (unsigned int)f2h(a) | ((unsigned int)f2h(b) << 16);
#endif
}

static __device__ __forceinline__ v8f wmma_f16(const Frag16& a, const Frag16& b, v8f c) {
  return __builtin_amdgcn_wmma_f32_16x16x32_f16(false, a.v, false, b.v,
                                                (short)0, c, false, false);
}

__global__ void __launch_bounds__(THREADS)
syn_fused_kernel(const float* __restrict__ x, const float* __restrict__ w,
                 const float* __restrict__ alpha, const float* __restrict__ beta,
                 float* __restrict__ out) {
  extern __shared__ __align__(16) char smem[];
  unsigned short* sW  = (unsigned short*)smem;                         // [OT][KP] f16, col-major
  unsigned short* sA0 = (unsigned short*)(smem + SW_BYTES);            // [BT][KP] f16, buf 0
  unsigned short* sA1 = (unsigned short*)(smem + SW_BYTES + SA_BYTES); // buf 1
  float*          sP  = (float*)(smem + SW_BYTES + 2 * SA_BYTES);      // [4][BT][PW] partials

  const int tid  = threadIdx.x;
  const int lane = tid & 31;
  const int wave = tid >> 5;    // 0..7
  const int kq   = wave >> 1;   // 0..3 : K quarter (k in [kq*256, kq*256+256))
  const int slot = wave & 1;    // 0..1 : N half (cols slot*32 .. +31)
  const int half = lane >> 4;   // 0/1
  const int n15  = lane & 15;

  const int oBase = blockIdx.x * OT;
  const int bBase = blockIdx.y * BT;

  // ---- stage W tile once: f32 -> f16, transposed to [o_local][k] ----
  for (int idx = tid; idx < (I_ * OT) / 4; idx += THREADS) {
    const int k  = idx >> 4;            // OT/4 = 16 float4 per k-row
    const int o4 = (idx & 15) << 2;
    const float4 wv = *reinterpret_cast<const float4*>(w + (size_t)k * O_ + oBase + o4);
    sW[(size_t)(o4 + 0) * KP + k] = f2h(wv.x);
    sW[(size_t)(o4 + 1) * KP + k] = f2h(wv.y);
    sW[(size_t)(o4 + 2) * KP + k] = f2h(wv.z);
    sW[(size_t)(o4 + 3) * KP + k] = f2h(wv.w);
  }

  // ---- stage x[:, t=0, :] into buffer 0 ----
  for (int idx = tid; idx < (BT * I_) / 4; idx += THREADS) {
    const int r  = idx >> 8;
    const int c4 = (idx & 255) << 2;
    const float4 xv = *reinterpret_cast<const float4*>(
        x + ((size_t)(bBase + r) * T_) * I_ + c4);
    uint2 p; p.x = pk2(xv.x, xv.y); p.y = pk2(xv.z, xv.w);
    *reinterpret_cast<uint2*>(sA0 + (size_t)r * KP + c4) = p;
  }

  // ---- per-lane reduce/scan mapping: 8 consecutive output cols ----
  const int mR   = wave * 4 + (lane >> 3);   // output row 0..31
  const int nloc = (lane & 7) << 3;          // output col chunk (8 wide)
  nf4 dec0, dec1, syn0, syn1;
  {
    const float4 a0 = *reinterpret_cast<const float4*>(alpha + oBase + nloc);
    const float4 a1 = *reinterpret_cast<const float4*>(alpha + oBase + nloc + 4);
    const float4 b0 = *reinterpret_cast<const float4*>(beta  + oBase + nloc);
    const float4 b1 = *reinterpret_cast<const float4*>(beta  + oBase + nloc + 4);
    dec0.x = a0.x * (1.0f - b0.x); dec0.y = a0.y * (1.0f - b0.y);
    dec0.z = a0.z * (1.0f - b0.z); dec0.w = a0.w * (1.0f - b0.w);
    dec1.x = a1.x * (1.0f - b1.x); dec1.y = a1.y * (1.0f - b1.y);
    dec1.z = a1.z * (1.0f - b1.z); dec1.w = a1.w * (1.0f - b1.w);
    syn0 = (nf4)0.0f;
    syn1 = (nf4)0.0f;
  }

  // syn[t=0] = 0 (coalesced, non-temporal: output is write-once streaming)
  {
    nf4 z = (nf4)0.0f;
    float* po = out + ((size_t)(bBase + mR) * T_) * O_ + oBase + nloc;
    __builtin_nontemporal_store(z, reinterpret_cast<nf4*>(po));
    __builtin_nontemporal_store(z, reinterpret_cast<nf4*>(po + 4));
  }

  // ---- GEMM fragment base pointers (ISA layouts, CDNA5 ISA §7.12.2) ----
  // this wave: K quarter kq, rows 0..31 (2 M tiles), cols slot*32..+31 (2 N tiles)
  const int kOff = kq * 256;
  const size_t aOff0 = (size_t)(n15)      * KP + kOff + half * 8;
  const size_t aOff1 = (size_t)(16 + n15) * KP + kOff + half * 8;
  const unsigned short* bBase0 = sW + (size_t)(slot * 32 + n15)      * KP + kOff + half * 8;
  const unsigned short* bBase1 = sW + (size_t)(slot * 32 + 16 + n15) * KP + kOff + half * 8;
  const int nW0 = slot * 32 + n15;        // partial-write cols
  const int nW1 = nW0 + 16;

  __syncthreads();   // W + A(t=0) staged

  for (int t = 0; t < T_ - 1; ++t) {
    const unsigned short* cur = (t & 1) ? sA1 : sA0;
    unsigned short*       nxt = (t & 1) ? sA0 : sA1;

    // ---- stage x[:, t+1, :] into the other buffer (overlaps with WMMA below) ----
    if (t < T_ - 2) {
      for (int idx = tid; idx < (BT * I_) / 4; idx += THREADS) {
        const int r  = idx >> 8;
        const int c4 = (idx & 255) << 2;
        const float4 xv = *reinterpret_cast<const float4*>(
            x + ((size_t)(bBase + r) * T_ + (t + 1)) * I_ + c4);
        uint2 p; p.x = pk2(xv.x, xv.y); p.y = pk2(xv.z, xv.w);
        *reinterpret_cast<uint2*>(nxt + (size_t)r * KP + c4) = p;
      }
      // hint-prefetch t+2 (global_prefetch_b8)
      if (t < T_ - 3) {
        const int pr = tid >> 3;
        const int pc = (tid & 7) << 7;
        __builtin_prefetch(x + ((size_t)(bBase + pr) * T_ + (t + 2)) * I_ + pc, 0, 0);
      }
    }

    // ---- 32x32 partial GEMM over this wave's K quarter: 2 ds_load_b128 / wmma ----
    v8f c00 = {}, c01 = {}, c10 = {}, c11 = {};
    const unsigned short* a0p = cur + aOff0;
    const unsigned short* a1p = cur + aOff1;
    #pragma unroll
    for (int kk = 0; kk < 8; ++kk) {
      Frag16 a0, a1, b0, b1;
      const uint4* pa0 = reinterpret_cast<const uint4*>(a0p + kk * 32);
      const uint4* pa1 = reinterpret_cast<const uint4*>(a1p + kk * 32);
      const uint4* pb0 = reinterpret_cast<const uint4*>(bBase0 + kk * 32);
      const uint4* pb1 = reinterpret_cast<const uint4*>(bBase1 + kk * 32);
      a0.q[0] = pa0[0]; a0.q[1] = pa0[2];   // k chunks +half*8, +16+half*8
      a1.q[0] = pa1[0]; a1.q[1] = pa1[2];
      b0.q[0] = pb0[0]; b0.q[1] = pb0[2];   // k chunks +half*8, +16+half*8
      b1.q[0] = pb1[0]; b1.q[1] = pb1[2];
      c00 = wmma_f16(a0, b0, c00);
      c01 = wmma_f16(a0, b1, c01);
      c10 = wmma_f16(a1, b0, c10);
      c11 = wmma_f16(a1, b1, c11);
    }
    __syncthreads();   // cur reads done (WAR) + nxt staging visible (RAW for next iter)

    // ---- write K-quarter partials to scratch ----
    {
      float* base = sP + (size_t)(kq * BT) * PW;
      #pragma unroll
      for (int r = 0; r < 8; ++r) {
        const int m0 = half * 8 + r;
        base[(size_t)m0 * PW + nW0]        = c00[r];
        base[(size_t)m0 * PW + nW1]        = c01[r];
        base[(size_t)(m0 + 16) * PW + nW0] = c10[r];
        base[(size_t)(m0 + 16) * PW + nW1] = c11[r];
      }
    }
    __syncthreads();   // partials visible

    // ---- reduce 4 K-quarters, apply recurrence, store syn at t+1 ----
    nf4 s0 = (nf4)0.0f, s1 = (nf4)0.0f;
    #pragma unroll
    for (int q = 0; q < 4; ++q) {
      const nf4* pp = reinterpret_cast<const nf4*>(
          sP + (size_t)(q * BT + mR) * PW + nloc);
      s0 += pp[0];
      s1 += pp[1];
    }
    syn0 = dec0 * syn0 + s0;
    syn1 = dec1 * syn1 + s1;
    float* po = out + ((size_t)(bBase + mR) * T_ + (t + 1)) * O_ + oBase + nloc;
    __builtin_nontemporal_store(syn0, reinterpret_cast<nf4*>(po));
    __builtin_nontemporal_store(syn1, reinterpret_cast<nf4*>(po + 4));
  }
}

extern "C" void kernel_launch(void* const* d_in, const int* in_sizes, int n_in,
                              void* d_out, int out_size, void* d_ws, size_t ws_size,
                              hipStream_t stream) {
  const float* x     = (const float*)d_in[0];
  const float* w     = (const float*)d_in[1];
  const float* alpha = (const float*)d_in[2];
  const float* beta  = (const float*)d_in[3];
  float* out = (float*)d_out;

  (void)hipFuncSetAttribute((const void*)syn_fused_kernel,
                            hipFuncAttributeMaxDynamicSharedMemorySize, (int)SMEM_BYTES);

  dim3 grid(O_ / OT, B_ / BT);   // (32, 8) = 256 workgroups
  dim3 block(THREADS);
  syn_fused_kernel<<<grid, block, SMEM_BYTES, stream>>>(x, w, alpha, beta, out);
}
